// SoftMaskedMultiheadAttention_64974265254267
// MI455X (gfx1250) — matrix operand order repaired and checked by reading, not compiled, not run on hardware
//
#include <hip/hip_runtime.h>
#include <hip/hip_bf16.h>

// Problem constants (match reference)
#define Bc 4
#define Lc 1024
#define Ec 1024
#define Hc 16
#define Dc 64
#define SCALEc 8.0f
#define EPSc 1e-6f

// scores are processed in the log2 domain:  t = log2e * (qk/sqrt(D) + bias)
// kC = log2e / sqrt(D); the mask bias is stored pre-multiplied by log2e.
#define LOG2E 1.4426950408889634f
#define kCf (0.125f * LOG2E)

typedef __attribute__((ext_vector_type(16))) __bf16 v16bf;
typedef __attribute__((ext_vector_type(8)))  float  v8f;
typedef __attribute__((ext_vector_type(4)))  float  v4f;
typedef __attribute__((ext_vector_type(4)))  unsigned short v4us;

// Raw hardware transcendentals (v_exp_f32 / v_log_f32, no denormal-range
// fixup chain -- probabilities below 2^-126 flush to zero, which is exactly
// what softmax wants).
static __device__ __forceinline__ float exp2_raw(float x) { return __builtin_amdgcn_exp2f(x); }
static __device__ __forceinline__ float log2_raw(float x) { return __builtin_amdgcn_logf(x); }

// ---------- bf16 helpers (round-to-nearest-even) ----------
static __device__ __forceinline__ unsigned short bf16bits(float f) {
    unsigned u = __builtin_bit_cast(unsigned, f);
    u += 0x7FFFu + ((u >> 16) & 1u);
    return (unsigned short)(u >> 16);
}

// K mapping for A-operand storage: identity (plain order).  A contiguous
// 16-element read at offset (laneHalf*16) yields K slots {0..15} for lanes
// 0-15 and {16..31} for lanes 16-31, matching the B-operand reads.
static __device__ __forceinline__ int permK(int k) {
    return k;
}

// ---------- unified 16x32 fragment loader (wave32) ----------
// One 32-byte load per lane:  p + (lane&15)*ld + (lane>>4)*16.
// Serves B fragments from plain "TN" storage and A fragments from permuted-K
// storage (see permK above).
static __device__ __forceinline__ v16bf load_frag32(const unsigned short* p, int ld, int lane) {
    const int l16 = lane & 15, lh = (lane >> 4) & 1;
    return *reinterpret_cast<const v16bf*>(p + l16 * ld + lh * 16);
}
static __device__ __forceinline__ v8f wmma_bf16(v16bf a, v16bf b, v8f c) {
    return __builtin_amdgcn_wmma_f32_16x16x32_bf16(false, a, false, b, (short)0, c, false, false);
}

// ---------- pre-pass kernels ----------
// Vectorized f32 -> bf16; PERM applies the A-operand K-permutation (permK
// preserves 4-element alignment, so the ushort4 store stays contiguous).
template <bool PERM>
__global__ __launch_bounds__(256) void f2bf_kernel(const float* __restrict__ in,
                                                   unsigned short* __restrict__ out, int n4) {
    int i = blockIdx.x * 256 + threadIdx.x;
    if (i >= n4) return;
    v4f x = reinterpret_cast<const v4f*>(in)[i];
    v4us y;
#pragma unroll
    for (int j = 0; j < 4; ++j) y[j] = bf16bits(x[j]);
    int idx = i * 4;
    *reinterpret_cast<v4us*>(out + (PERM ? permK(idx) : idx)) = y;
}
// colbias (log2 domain) = key_padding ? -inf : log2e * SCALE * ln(clip(mask))
//                       = key_padding ? -inf : SCALE * log2(clip(mask, EPS))
__global__ __launch_bounds__(256) void bias_kernel(const float* __restrict__ mask,
                                                   const unsigned char* __restrict__ kpm,
                                                   float* __restrict__ cb, int n) {
    int i = blockIdx.x * 256 + threadIdx.x;
    if (i < n) cb[i] = kpm[i] ? -INFINITY : SCALEc * log2_raw(fmaxf(mask[i], EPSc));
}

// ---------- tiled WMMA GEMM: out = A(4096x1024,bf16 permK) @ W^T + bias ----------
// Block tile 128(M) x 128(N): 8 waves as 4(M) x 2(N), each wave 2x4 fragments
// -> 6 fragment loads per 8 WMMAs.
// MODE 0: out bf16 (B,H,L,D) plain        (K projection -> B operand of scores)
// MODE 1: out bf16 (B,H,D,L) transposed   (V projection -> B operand of P@V)
// MODE 2: out f32 (B,L,E) row-major       (final output projection)
// MODE 3: out bf16 (B,H,L,D) permK on D   (Q projection -> A operand of scores)
template <int MODE>
__global__ __launch_bounds__(256)
void gemm128x128(const unsigned short* __restrict__ A, const unsigned short* __restrict__ Wb,
                 const float* __restrict__ bias, void* __restrict__ outp) {
    const int lane = threadIdx.x & 31, wave = threadIdx.x >> 5;
    const int m0 = blockIdx.y * 128 + (wave & 3) * 32;   // 32 blocks * 128 = 4096
    const int n0 = blockIdx.x * 128 + (wave >> 2) * 64;  //  8 blocks * 128 = 1024
    const int l16 = lane & 15, lh = lane >> 4;

    v8f acc[2][4] = {};
#pragma unroll 2
    for (int k0 = 0; k0 < Ec; k0 += 32) {
        __builtin_prefetch(Wb + (n0 + l16) * Ec + k0 + 32, 0, 3);  // global_prefetch_b8
        v16bf a0 = load_frag32(A + (m0 + 0) * Ec + k0, Ec, lane);
        v16bf a1 = load_frag32(A + (m0 + 16) * Ec + k0, Ec, lane);
        v16bf b0 = load_frag32(Wb + (n0 + 0) * Ec + k0, Ec, lane);
        v16bf b1 = load_frag32(Wb + (n0 + 16) * Ec + k0, Ec, lane);
        v16bf b2 = load_frag32(Wb + (n0 + 32) * Ec + k0, Ec, lane);
        v16bf b3 = load_frag32(Wb + (n0 + 48) * Ec + k0, Ec, lane);
        acc[0][0] = wmma_bf16(a0, b0, acc[0][0]);
        acc[0][1] = wmma_bf16(a0, b1, acc[0][1]);
        acc[0][2] = wmma_bf16(a0, b2, acc[0][2]);
        acc[0][3] = wmma_bf16(a0, b3, acc[0][3]);
        acc[1][0] = wmma_bf16(a1, b0, acc[1][0]);
        acc[1][1] = wmma_bf16(a1, b1, acc[1][1]);
        acc[1][2] = wmma_bf16(a1, b2, acc[1][2]);
        acc[1][3] = wmma_bf16(a1, b3, acc[1][3]);
    }
#pragma unroll
    for (int i = 0; i < 2; ++i)
#pragma unroll
        for (int j = 0; j < 4; ++j)
#pragma unroll
            for (int r = 0; r < 8; ++r) {
                int row = m0 + i * 16 + r + lh * 8;   // C layout: VGPR r -> M = r (+8 hi half)
                int col = n0 + j * 16 + l16;          // N = lane&15
                float v = acc[i][j][r] + bias[col];
                if (MODE == 2) {
                    ((float*)outp)[(long)row * Ec + col] = v;
                } else {
                    int bb = row >> 10, tok = row & (Lc - 1);
                    int hh = col >> 6, dd = col & (Dc - 1);
                    unsigned short* ob = (unsigned short*)outp;
                    if (MODE == 0)
                        ob[(((long)(bb * Hc + hh)) * Lc + tok) * Dc + dd] = bf16bits(v);
                    else if (MODE == 3)
                        ob[(((long)(bb * Hc + hh)) * Lc + tok) * Dc + permK(dd)] = bf16bits(v);
                    else
                        ob[(((long)(bb * Hc + hh)) * Dc + dd) * Lc + tok] = bf16bits(v);
                }
            }
}

// ---------- attention: scores -> two-pass softmax -> P@V, plus head-mean ----------
// grid (L/16, B), 256 threads = 8 waves, each wave does heads w and w+8 for one
// 16-row query tile.  Dynamic LDS: 64KB f32 mean tile + 8KB bf16 P staging.
// Softmax runs in the log2 domain with PER-LANE online stats (one raw v_exp_f32
// per element, select-based max update -> no canonicalize chains); a single
// butterfly per head merges the 16 column groups; pass 2 folds normalization
// into the exponent.
__global__ __launch_bounds__(256)
void attn_kernel(const unsigned short* __restrict__ Qb, const unsigned short* __restrict__ Kb,
                 const unsigned short* __restrict__ Vt, const float* __restrict__ colbias,
                 unsigned short* __restrict__ ctx, float* __restrict__ meanOut) {
    extern __shared__ char smem[];
    float* meanP = (float*)smem;                                   // [16][1024]
    unsigned short* pst = (unsigned short*)(smem + 16 * Lc * 4);   // 8 waves * 16*32

    const int tid = threadIdx.x;
    const int lane = tid & 31, wave = tid >> 5;
    const int l16 = lane & 15, lh = lane >> 4;
    const int b = blockIdx.y;
    const int q0 = blockIdx.x * 16;

    for (int i = tid; i < 16 * Lc; i += 256) meanP[i] = 0.f;
    __syncthreads();

    unsigned short* mypst = pst + wave * (16 * 32);
    // P-stage store columns (permuted-K A layout), constant per lane
    const int posLo = permK(l16);        // key (j0 + l16)
    const int posHi = permK(16 + l16);   // key (j0 + 16 + l16)

    for (int hrep = 0; hrep < 2; ++hrep) {
        const int h = wave + hrep * 8;
        const long hb = (long)(b * Hc + h);
        const unsigned short* Qh = Qb + hb * Lc * Dc;   // permK over D
        const unsigned short* Kh = Kb + hb * Lc * Dc;   // plain
        const unsigned short* Vh = Vt + hb * Dc * Lc;   // transposed (D,L)
        const float* cb = colbias + b * Lc;

        // Q fragments for this 16-row tile, D=64 split into two K=32 chunks
        v16bf qa0 = load_frag32(Qh + q0 * Dc + 0, Dc, lane);
        v16bf qa1 = load_frag32(Qh + q0 * Dc + 32, Dc, lane);

        float m2[8], lsum[8];
#pragma unroll
        for (int r = 0; r < 8; ++r) { m2[r] = -1e30f; lsum[r] = 0.f; }

        // ---- pass 1: per-lane online log2-domain stats (1 raw exp2 per element) ----
        for (int j0 = 0; j0 < Lc; j0 += 16) {
            v16bf kb0 = load_frag32(Kh + j0 * Dc + 0, Dc, lane);
            v16bf kb1 = load_frag32(Kh + j0 * Dc + 32, Dc, lane);
            v8f c = {};
            c = wmma_bf16(qa0, kb0, c);
            c = wmma_bf16(qa1, kb1, c);
            float bias2 = cb[j0 + l16];
#pragma unroll
            for (int r = 0; r < 8; ++r) {
                float t = fmaf(c[r], kCf, bias2);            // log2-domain score
                float d = t - m2[r];
                float e = exp2_raw(-__builtin_fabsf(d));     // v_exp_f32 with -|d| modifier
                bool  g = d > 0.f;
                lsum[r] = fmaf(lsum[r], g ? e : 1.f, g ? 1.f : e);
                m2[r]   = g ? t : m2[r];                     // select, no canonicalize max
            }
        }
        // ---- single cross-lane merge of the 16 column groups ----
        float M2[8];
#pragma unroll
        for (int r = 0; r < 8; ++r) M2[r] = m2[r];
#pragma unroll
        for (int d = 1; d < 16; d <<= 1)
#pragma unroll
            for (int r = 0; r < 8; ++r) M2[r] = fmaxf(M2[r], __shfl_xor(M2[r], d, 16));
#pragma unroll
        for (int r = 0; r < 8; ++r) lsum[r] *= exp2_raw(m2[r] - M2[r]);
#pragma unroll
        for (int d = 1; d < 16; d <<= 1)
#pragma unroll
            for (int r = 0; r < 8; ++r) lsum[r] += __shfl_xor(lsum[r], d, 16);
        // z = log2 of the softmax denominator in shifted domain
        float z[8];
#pragma unroll
        for (int r = 0; r < 8; ++r)
            z[r] = (lsum[r] > 0.f) ? (M2[r] + log2_raw(lsum[r])) : 3.0e38f;

        // ---- pass 2: normalized P = exp2(t - z), mean accumulation, P@V ----
        v8f o[4] = {};
        for (int j0 = 0; j0 < Lc; j0 += 32) {
#pragma unroll
            for (int half = 0; half < 2; ++half) {
                const int jj = j0 + half * 16;
                v16bf kb0 = load_frag32(Kh + jj * Dc + 0, Dc, lane);
                v16bf kb1 = load_frag32(Kh + jj * Dc + 32, Dc, lane);
                v8f c = {};
                c = wmma_bf16(qa0, kb0, c);
                c = wmma_bf16(qa1, kb1, c);
                float bias2 = cb[jj + l16];
                const int pos = half ? posHi : posLo;
#pragma unroll
                for (int r = 0; r < 8; ++r) {
                    float pn = exp2_raw(fmaf(c[r], kCf, bias2) - z[r]);  // normalized prob
                    int row = r + lh * 8;
                    atomicAdd(&meanP[row * Lc + jj + l16], pn * (1.0f / Hc));  // ds_add_f32
                    mypst[row * 32 + pos] = bf16bits(pn);
                }
            }
            asm volatile("s_wait_dscnt 0" ::: "memory");   // LDS P tile visible to reads
            v16bf pa = load_frag32(mypst, 32, lane);       // P as A fragment (16x32, permK)
#pragma unroll
            for (int f = 0; f < 4; ++f) {
                v16bf vb = load_frag32(Vh + (long)(f * 16) * Lc + j0, Lc, lane);  // V^T rows
                o[f] = wmma_bf16(pa, vb, o[f]);
            }
        }
        // store per-head context (already normalized) into (B,L,E) bf16, permK
        // over E so it can serve as A operand of the output projection
#pragma unroll
        for (int f = 0; f < 4; ++f)
#pragma unroll
            for (int r = 0; r < 8; ++r) {
                int row = q0 + r + lh * 8;
                int col = h * Dc + permK(f * 16 + l16);
                ctx[((long)(b * Lc) + row) * Ec + col] = bf16bits(o[f][r]);
            }
    }
    __syncthreads();
    for (int i = tid; i < 16 * Lc; i += 256) {
        int row = i >> 10, col = i & (Lc - 1);
        meanOut[((long)(b * Lc) + q0 + row) * Lc + col] = meanP[i];
    }
}

// ---------- host launch ----------
extern "C" void kernel_launch(void* const* d_in, const int* in_sizes, int n_in,
                              void* d_out, int out_size, void* d_ws, size_t ws_size,
                              hipStream_t stream) {
    (void)in_sizes; (void)n_in; (void)out_size; (void)ws_size;
    const float* query = (const float*)d_in[0];
    const float* key   = (const float*)d_in[1];
    const float* value = (const float*)d_in[2];
    const float* amask = (const float*)d_in[3];
    const unsigned char* kpm = (const unsigned char*)d_in[4];
    const float* Wq = (const float*)d_in[5];  const float* bq = (const float*)d_in[6];
    const float* Wk = (const float*)d_in[7];  const float* bk = (const float*)d_in[8];
    const float* Wv = (const float*)d_in[9];  const float* bv = (const float*)d_in[10];
    const float* Wo = (const float*)d_in[11]; const float* bo = (const float*)d_in[12];

    float* out     = (float*)d_out;                       // (B,L,E)
    float* meanOut = out + (size_t)Bc * Lc * Ec;          // (B,L,L)

    char* ws = (char*)d_ws;
    const size_t MB = 1u << 20;
    unsigned short* Qb  = (unsigned short*)(ws + 0 * MB);   // (B,H,L,D) bf16 permK, 8MB
    unsigned short* Kb  = (unsigned short*)(ws + 8 * MB);   // (B,H,L,D) bf16 plain, 8MB
    unsigned short* Vt  = (unsigned short*)(ws + 16 * MB);  // (B,H,D,L) bf16 plain, 8MB
    unsigned short* ctx = (unsigned short*)(ws + 24 * MB);  // (B,L,E)   bf16 permK, 8MB
    unsigned short* WqB = (unsigned short*)(ws + 32 * MB);  // 2MB each, plain
    unsigned short* WkB = (unsigned short*)(ws + 34 * MB);
    unsigned short* WvB = (unsigned short*)(ws + 36 * MB);
    unsigned short* WoB = (unsigned short*)(ws + 38 * MB);
    unsigned short* Xq  = (unsigned short*)(ws + 40 * MB);  // (B*L,E) bf16 permK, 8MB
    unsigned short* Xk  = (unsigned short*)(ws + 48 * MB);
    unsigned short* Xv  = (unsigned short*)(ws + 56 * MB);
    float* colbias      = (float*)(ws + 64 * MB);           // (B,L) f32, log2 domain

    const int nW4 = Ec * Ec / 4;          // weight quads
    const int nX4 = Bc * Lc * Ec / 4;     // activation quads
    f2bf_kernel<false><<<(nW4 + 255) / 256, 256, 0, stream>>>(Wq, WqB, nW4);
    f2bf_kernel<false><<<(nW4 + 255) / 256, 256, 0, stream>>>(Wk, WkB, nW4);
    f2bf_kernel<false><<<(nW4 + 255) / 256, 256, 0, stream>>>(Wv, WvB, nW4);
    f2bf_kernel<false><<<(nW4 + 255) / 256, 256, 0, stream>>>(Wo, WoB, nW4);
    f2bf_kernel<true><<<(nX4 + 255) / 256, 256, 0, stream>>>(query, Xq, nX4);
    f2bf_kernel<true><<<(nX4 + 255) / 256, 256, 0, stream>>>(key,   Xk, nX4);
    f2bf_kernel<true><<<(nX4 + 255) / 256, 256, 0, stream>>>(value, Xv, nX4);
    bias_kernel<<<(Bc * Lc + 255) / 256, 256, 0, stream>>>(amask, kpm, colbias, Bc * Lc);

    dim3 gg(Ec / 128, (Bc * Lc) / 128);  // (8, 32)
    gemm128x128<3><<<gg, 256, 0, stream>>>(Xq, WqB, bq, Qb);
    gemm128x128<0><<<gg, 256, 0, stream>>>(Xk, WkB, bk, Kb);
    gemm128x128<1><<<gg, 256, 0, stream>>>(Xv, WvB, bv, Vt);

    dim3 ga(Lc / 16, Bc);               // (64, 4)
    const size_t smem = 16 * Lc * sizeof(float) + 8 * 16 * 32 * sizeof(unsigned short);
    attn_kernel<<<ga, 256, smem, stream>>>(Qb, Kb, Vt, colbias, ctx, meanOut);

    gemm128x128<2><<<gg, 256, 0, stream>>>(ctx, WoB, bo, out);
}
